// BTSPMemoryBank_37031208026221
// MI455X (gfx1250) — matrix operand order repaired and checked by reading, not compiled, not run on hardware
//
#include <hip/hip_runtime.h>
#include <hip/hip_bf16.h>

typedef __attribute__((ext_vector_type(16))) _Float16 v16h;
typedef __attribute__((ext_vector_type(8)))  float    v8f;
typedef unsigned short u16;

#define D_DIM   256
#define TILE_B  16
#define CHUNK_N 8192
#define ITER_N  128     // 8 waves * 16 columns per block iteration
#define TOPK    32
#define GMAX    4       // max sub-scanners per row

// ---------------------------------------------------------------------------
// Kernel 0: q_proj = query @ W^T, L2-normalize rows, emit f16.
// One block per query row, 256 threads (one per output dim).
// W (256KB) stays L2-resident across the 1024 blocks.
// ---------------------------------------------------------------------------
__global__ __launch_bounds__(256) void qproj_kernel(
    const float* __restrict__ query, const float* __restrict__ W,
    u16* __restrict__ q_h)
{
    __shared__ float qrow[D_DIM];
    __shared__ float red[256];
    const int row = blockIdx.x;
    const int t   = threadIdx.x;

    qrow[t] = query[(size_t)row * D_DIM + t];
    __syncthreads();

    float acc = 0.f;
    const float* wr = W + (size_t)t * D_DIM;   // proj[i] = sum_j q[j]*W[i][j]
    #pragma unroll 8
    for (int j = 0; j < D_DIM; ++j) acc = fmaf(qrow[j], wr[j], acc);

    red[t] = acc * acc;
    __syncthreads();
    for (int s = 128; s > 0; s >>= 1) {
        if (t < s) red[t] += red[t + s];
        __syncthreads();
    }
    const float inv = 1.0f / fmaxf(sqrtf(red[0]), 1e-12f);
    const _Float16 hv = (_Float16)(acc * inv);
    q_h[(size_t)row * D_DIM + t] = __builtin_bit_cast(u16, hv);
}

// ---------------------------------------------------------------------------
// Kernel 1: fused WMMA GEMM (sim = q_normed @ memory^T) + per-chunk top-32.
// grid = (B/16, N/CHUNK_N); block = 256 threads = 8 wave32s.
// Each wave computes a 16x16 sim tile via 8 x v_wmma_f32_16x16x32_f16.
// Top-k: G sub-scanners per row, each owning a disjoint column slice and a
// private running top-32 (scan depth per iteration = ITER_N / G).
// ---------------------------------------------------------------------------
__global__ __launch_bounds__(256) void sim_topk_kernel(
    const u16* __restrict__ q_h, const float* __restrict__ memory,
    float* __restrict__ cand_v, int* __restrict__ cand_i,
    int nchunks, int G)
{
    __shared__ u16   qA[TILE_B][D_DIM + 8];      // +8 u16 keeps 16B align, breaks bank conflicts
    __shared__ float simL[TILE_B][ITER_N + 1];
    __shared__ float topv[TILE_B][GMAX][TOPK];   // [row][sub][k]
    __shared__ int   topi[TILE_B][GMAX][TOPK];

    const int tid     = threadIdx.x;
    const int rowBase = blockIdx.x * TILE_B;
    const int chunk   = blockIdx.y;
    const int nBase   = chunk * CHUNK_N;

    // Load q tile (16 rows x 256 f16) into LDS: 512 x 8-halfword copies.
    for (int it = 0; it < 2; ++it) {
        const int e = (tid + it * 256) * 8;      // element index, 0..4095
        const int r = e >> 8, c = e & 255;
        *(uint4*)&qA[r][c] = *(const uint4*)&q_h[(size_t)(rowBase + r) * D_DIM + c];
    }
    // Init running top-k state (all GMAX slots; unused ones stay -inf).
    for (int e = tid; e < TILE_B * GMAX * TOPK; e += 256) {
        ((float*)topv)[e] = -__builtin_inff();
        ((int*)topi)[e]   = 0;
    }
    __syncthreads();

    const int lane   = tid & 31;
    const int waveId = tid >> 5;
    const int mrow   = lane & 15;
    const int half   = lane >> 4;

    // A-matrix registers per ISA 16-bit 16x32 layout:
    // lane holds row M=lane%16; words 0-3 = K[8h .. 8h+7], words 4-7 = K[16+8h .. 16+8h+7].
    v16h a[8];
    #pragma unroll
    for (int ks = 0; ks < 8; ++ks) {
        const int kb = ks * 32 + half * 8;
        union { v16h v; uint4 u[2]; } ua;
        ua.u[0] = *(const uint4*)&qA[mrow][kb];
        ua.u[1] = *(const uint4*)&qA[mrow][kb + 16];
        a[ks] = ua.v;
    }

    // Sub-scanner identity: threads 0 .. 16*G-1, row = t/G, sub = t%G.
    const bool isScan = (tid < TILE_B * G);
    const int  sRow   = isScan ? (tid / G) : 0;
    const int  sSub   = isScan ? (tid - sRow * G) : 0;
    const int  seg    = ITER_N / G;              // columns per sub-scanner per iter
    float tmin = -__builtin_inff();
    int   tminpos = 0;

    const int iters = CHUNK_N / ITER_N;          // 64

    for (int it = 0; it < iters; ++it) {
        // B-matrix per ISA 32x16 16-bit layout: lane holds column N=lane%16
        // (a memory row), e 0-7 = K[8h .. 8h+7], e 8-15 = K[16+8h .. 16+8h+7] within each 32-K step.
        const int n = nBase + it * ITER_N + waveId * 16 + mrow;
        const float* mp = memory + (size_t)n * D_DIM;
        if (it + 1 < iters) __builtin_prefetch(mp + (size_t)ITER_N * D_DIM, 0, 0);

        v8f c = {0.f, 0.f, 0.f, 0.f, 0.f, 0.f, 0.f, 0.f};
        #pragma unroll
        for (int ks = 0; ks < 8; ++ks) {
            const int k0 = ks * 32 + half * 8;
            union { float4 q[4]; float f[16]; } uf;
            uf.q[0] = *(const float4*)(mp + k0);
            uf.q[1] = *(const float4*)(mp + k0 + 4);
            uf.q[2] = *(const float4*)(mp + k0 + 16);
            uf.q[3] = *(const float4*)(mp + k0 + 20);
            union { v16h v; _Float16 h[16]; } ub;
            #pragma unroll
            for (int e = 0; e < 16; ++e) ub.h[e] = (_Float16)uf.f[e];
            c = __builtin_amdgcn_wmma_f32_16x16x32_f16(
                    false, a[ks], false, ub.v, (short)0, c, false, false);
        }

        // C layout: VGPR r of lane -> M = r + 8*half, N = lane%16.
        #pragma unroll
        for (int r = 0; r < 8; ++r)
            simL[r + half * 8][waveId * 16 + mrow] = c[r];
        __syncthreads();

        // Running top-32: each sub-scanner scans its disjoint column slice.
        if (isScan) {
            const int jBase = sSub * seg;
            const int gBase = nBase + it * ITER_N;
            for (int j = jBase; j < jBase + seg; ++j) {
                const float v = simL[sRow][j];
                if (v > tmin) {
                    topv[sRow][sSub][tminpos] = v;
                    topi[sRow][sSub][tminpos] = gBase + j;
                    float m = topv[sRow][sSub][0]; int mp2 = 0;
                    #pragma unroll
                    for (int q = 1; q < TOPK; ++q) {
                        const float tv = topv[sRow][sSub][q];
                        if (tv < m) { m = tv; mp2 = q; }
                    }
                    tmin = m; tminpos = mp2;
                }
            }
        }
        __syncthreads();
    }

    // Emit this chunk's G*32 candidates per row.
    // LDS flat offset within a row = sub*TOPK + k, row stride GMAX*TOPK.
    const int perRow = G * TOPK;
    for (int e = tid; e < TILE_B * perRow; e += 256) {
        const int row = e / perRow, rem = e - row * perRow;
        const size_t g = (size_t)(rowBase + row) * nchunks + chunk;
        cand_v[g * perRow + rem] = ((float*)topv)[row * (GMAX * TOPK) + rem];
        cand_i[g * perRow + rem] = ((int*)topi)[row * (GMAX * TOPK) + rem];
    }
}

// ---------------------------------------------------------------------------
// Kernel 2: per-row global top-32 (sorted desc via 32 argmax rounds; all
// candidate sub-lists cover disjoint index sets, so no duplicates),
// softmax, weighted gather.
// ---------------------------------------------------------------------------
__global__ __launch_bounds__(256) void reduce_out_kernel(
    const float* __restrict__ cand_v, const int* __restrict__ cand_i,
    const float* __restrict__ memory,
    float* __restrict__ out_ret, float* __restrict__ out_sim,
    int nchunks, int G)
{
    const int row = blockIdx.x;
    const int t   = threadIdx.x;
    const int candPerRow = nchunks * G * TOPK;   // up to 4096
    const int C = (candPerRow + 255) / 256;      // up to 16

    float cv[16];
    int   ci[16];
    for (int j = 0; j < C; ++j) {
        const int p = t * C + j;
        if (p < candPerRow) {
            cv[j] = cand_v[(size_t)row * candPerRow + p];
            ci[j] = cand_i[(size_t)row * candPerRow + p];
        } else { cv[j] = -__builtin_inff(); ci[j] = 0; }
    }

    __shared__ float redv[256];
    __shared__ int   redp[256];
    __shared__ float topValL[TOPK];
    __shared__ int   topIdxL[TOPK];
    __shared__ float wts[TOPK];
    __shared__ float invSum;

    for (int round = 0; round < TOPK; ++round) {
        float mv = cv[0]; int ms = 0;
        for (int j = 1; j < C; ++j) if (cv[j] > mv) { mv = cv[j]; ms = j; }
        redv[t] = mv; redp[t] = (t << 4) | ms;
        __syncthreads();
        for (int s = 128; s > 0; s >>= 1) {
            if (t < s && redv[t + s] > redv[t]) {
                redv[t] = redv[t + s]; redp[t] = redp[t + s];
            }
            __syncthreads();
        }
        const int wp = redp[0];
        if (t == (wp >> 4)) {
            topValL[round] = redv[0];
            topIdxL[round] = ci[wp & 15];
            cv[wp & 15] = -__builtin_inff();
        }
        __syncthreads();
    }

    if (t < TOPK) wts[t] = __expf(topValL[t] - topValL[0]);  // [0] is the max
    __syncthreads();
    if (t == 0) {
        float s = 0.f;
        for (int k = 0; k < TOPK; ++k) s += wts[k];
        invSum = 1.0f / s;
    }
    __syncthreads();
    const float inv = invSum;

    float acc = 0.f;
    for (int k = 0; k < TOPK; ++k)
        acc = fmaf(wts[k] * inv, memory[(size_t)topIdxL[k] * D_DIM + t], acc);
    out_ret[(size_t)row * D_DIM + t] = acc;
    if (t < TOPK) out_sim[(size_t)row * TOPK + t] = topValL[t];
}

// ---------------------------------------------------------------------------
extern "C" void kernel_launch(void* const* d_in, const int* in_sizes, int n_in,
                              void* d_out, int out_size, void* d_ws, size_t ws_size,
                              hipStream_t stream) {
    const float* query  = (const float*)d_in[0];
    const float* memory = (const float*)d_in[1];
    const float* W_key  = (const float*)d_in[2];
    // d_in[3] = top_k scalar; fixed at 32 per setup.

    const int B = in_sizes[0] / D_DIM;           // 1024
    const int N = in_sizes[1] / D_DIM;           // 262144
    const int nchunks = N / CHUNK_N;             // 32

    // Pick the largest G in {4,2,1} whose workspace footprint fits ws_size.
    size_t qBytes = ((size_t)B * D_DIM * sizeof(u16) + 255) & ~(size_t)255;
    int G = 1;
    for (int g = GMAX; g >= 1; g >>= 1) {
        size_t need = qBytes + (size_t)B * nchunks * g * TOPK * (sizeof(float) + sizeof(int));
        if (need <= ws_size) { G = g; break; }
    }

    // Workspace layout: q_f16 | cand_vals | cand_idx
    char* ws = (char*)d_ws;
    u16* q_h = (u16*)ws;
    size_t off = qBytes;
    float* cand_v = (float*)(ws + off);
    off += (size_t)B * nchunks * G * TOPK * sizeof(float);
    int* cand_i = (int*)(ws + off);

    float* out_ret = (float*)d_out;
    float* out_sim = out_ret + (size_t)B * D_DIM;

    qproj_kernel<<<B, 256, 0, stream>>>(query, W_key, q_h);
    dim3 g1(B / TILE_B, nchunks);                // x fastest -> chunk stays L2-hot
    sim_topk_kernel<<<g1, 256, 0, stream>>>(q_h, memory, cand_v, cand_i, nchunks, G);
    reduce_out_kernel<<<B, 256, 0, stream>>>(cand_v, cand_i, memory,
                                             out_ret, out_sim, nchunks, G);
}
